// MultiHeadAttention_19507741458712
// MI455X (gfx1250) — matrix operand order repaired and checked by reading, not compiled, not run on hardware
//
#include <hip/hip_runtime.h>

// ---------------------------------------------------------------------------
// MHA forward for B=2, S=2048, D=1024, H=16, HD=64 on gfx1250 (wave32, WMMA)
// ---------------------------------------------------------------------------

typedef __attribute__((ext_vector_type(16))) _Float16 v16h;
typedef __attribute__((ext_vector_type(8)))  _Float16 v8h;
typedef __attribute__((ext_vector_type(4)))  _Float16 v4h;
typedef __attribute__((ext_vector_type(8)))  float    v8f;

#define WMMA_F16(a, b, c) \
  __builtin_amdgcn_wmma_f32_16x16x32_f16(false, (a), false, (b), (short)0, (c), false, false)

// --- async global->LDS 16B copy (ASYNCcnt path, ISA 10 / 15.18.3) -----------
// vdst VGPR carries the LDS byte address (flat aperture: low 32 bits), vaddr
// is the 64-bit global address.  Tracked by ASYNCcnt, not LOADcnt/DScnt.
__device__ __forceinline__ void async_copy_b128(const _Float16* gptr,
                                                const _Float16* lptr) {
  unsigned long long ga = (unsigned long long)(uintptr_t)gptr;
  unsigned int la = (unsigned int)(uintptr_t)lptr;
  asm volatile("global_load_async_to_lds_b128 %0, %1, off"
               :: "v"(la), "v"(ga) : "memory");
}
__device__ __forceinline__ void wait_async0() {
  asm volatile("s_wait_asynccnt 0x0" ::: "memory");
}

// --- fragment loaders (layouts per CDNA5 ISA 7.12.2, wave32) ----------------
// A-matrix 16x32 f16, row-major source with row stride `ld` halves.
__device__ __forceinline__ v16h load_a_frag(const _Float16* base, int ld) {
  const int lane = threadIdx.x & 31;
  const int row = lane & 15, hi = lane >> 4;
  const _Float16* p = base + row * ld + hi * 8;
  v8h lo = *(const v8h*)(p);
  v8h hh = *(const v8h*)(p + 16);
  v16h a;
#pragma unroll
  for (int i = 0; i < 8; ++i) { a[i] = lo[i]; a[i + 8] = hh[i]; }
  return a;
}

// B-matrix 32x16 f16 stored [N][K] with row stride `ld` halves.
__device__ __forceinline__ v16h load_b_frag(const _Float16* base, int ld) {
  const int lane = threadIdx.x & 31;
  const int col = lane & 15, hi = lane >> 4;
  const _Float16* p = base + col * ld + hi * 8;
  v8h lo = *(const v8h*)(p);
  v8h hh = *(const v8h*)(p + 16);
  v16h b;
#pragma unroll
  for (int i = 0; i < 8; ++i) { b[i] = lo[i]; b[i + 8] = hh[i]; }
  return b;
}

// ---------------------------------------------------------------------------
// Kernel 1: QKV projection.  out(f16) = x(f32)[4096,1024] @ W(f32)[1024,1024] + b
// transposed==0 : write [B,H,S,HD]   (Q, K)
// transposed==1 : write [B,H,HD,S]   (V, pre-transposed for P@V B-fragments)
// ---------------------------------------------------------------------------
__global__ void __launch_bounds__(256)
qkv_gemm(const float* __restrict__ X, const float* __restrict__ W,
         const float* __restrict__ bias, _Float16* __restrict__ out,
         int transposed) {
  __shared__ _Float16 As[128 * 40];  // [128][32] + pad, row stride 80B
  __shared__ _Float16 Bt[128 * 40];  // W tile transposed: [N=128][K=32] + pad
  const int tid = threadIdx.x;
  const int wave = tid >> 5, lane = tid & 31;
  const int waveM = wave >> 2, waveN = wave & 3;  // 2 x 4 wave grid
  const int l = lane & 15, hi = lane >> 4;
  const int m0 = blockIdx.x * 128;
  const int n0 = blockIdx.y * 128;
  const int K = 1024, N = 1024;

  v8f acc[4][2];
#pragma unroll
  for (int mf = 0; mf < 4; ++mf)
#pragma unroll
    for (int nf = 0; nf < 2; ++nf) acc[mf][nf] = (v8f)(0.0f);

  for (int k0 = 0; k0 < K; k0 += 32) {
    __syncthreads();
    // stage A tile 128x32: float4 loads, f32->f16, 8B DS stores
#pragma unroll
    for (int i = 0; i < 4; ++i) {
      int idx = tid + i * 256;            // 0..1023 -> 128 rows x 8 float4
      int r = idx >> 3, c4 = idx & 7;
      float4 f = *(const float4*)&X[(size_t)(m0 + r) * K + k0 + c4 * 4];
      v4h h4 = {(_Float16)f.x, (_Float16)f.y, (_Float16)f.z, (_Float16)f.w};
      *(v4h*)&As[r * 40 + c4 * 4] = h4;
    }
    // stage W tile 32x128 transposed to [N][K]: float4 loads, scatter stores
#pragma unroll
    for (int i = 0; i < 4; ++i) {
      int idx = tid + i * 256;            // 0..1023 -> 32 rows x 32 float4
      int r = idx >> 5, c4 = idx & 31;
      float4 f = *(const float4*)&W[(size_t)(k0 + r) * N + n0 + c4 * 4];
      Bt[(c4 * 4 + 0) * 40 + r] = (_Float16)f.x;
      Bt[(c4 * 4 + 1) * 40 + r] = (_Float16)f.y;
      Bt[(c4 * 4 + 2) * 40 + r] = (_Float16)f.z;
      Bt[(c4 * 4 + 3) * 40 + r] = (_Float16)f.w;
    }
    __syncthreads();

    v16h bf[2];
#pragma unroll
    for (int nf = 0; nf < 2; ++nf)
      bf[nf] = load_b_frag(&Bt[(waveN * 32 + nf * 16) * 40], 40);
#pragma unroll
    for (int mf = 0; mf < 4; ++mf) {
      v16h af = load_a_frag(&As[(waveM * 64 + mf * 16) * 40], 40);
#pragma unroll
      for (int nf = 0; nf < 2; ++nf)
        acc[mf][nf] = WMMA_F16(af, bf[nf], acc[mf][nf]);
    }
  }

  // epilogue: bias add, convert, scatter into attention-friendly layout
#pragma unroll
  for (int mf = 0; mf < 4; ++mf)
#pragma unroll
    for (int nf = 0; nf < 2; ++nf)
#pragma unroll
      for (int r = 0; r < 8; ++r) {
        int m = m0 + waveM * 64 + mf * 16 + r + 8 * hi;  // global row (b*2048+s)
        int n = n0 + waveN * 32 + nf * 16 + l;           // output feature
        float v = acc[mf][nf][r] + bias[n];
        int bb = m >> 11, s = m & 2047;
        int h = n >> 6, hd = n & 63;
        size_t off = transposed
            ? (((size_t)(bb * 16 + h) * 64 + hd) * 2048 + s)
            : (((size_t)(bb * 16 + h) * 2048 + s) * 64 + hd);
        out[off] = (_Float16)v;
      }
}

// ---------------------------------------------------------------------------
// Kernel 2: flash attention.  One block = 64 query rows of one (b,h).
// Q,K: [B,H,S,64] f16 ; Vt: [B,H,64,S] f16 ; ctx out: [B,S,1024] f16
// ---------------------------------------------------------------------------
__global__ void __launch_bounds__(128)
attn_kernel(const _Float16* __restrict__ Q, const _Float16* __restrict__ Kg,
            const _Float16* __restrict__ Vt, const float* __restrict__ mask,
            _Float16* __restrict__ ctx) {
  __shared__ _Float16 Ks[64 * 72];       // key tile  [key][hd]  == [N][K] for scores
  __shared__ _Float16 Vs[64 * 72];       // V^T tile  [hd][key]  == [N][K] for P@V
  __shared__ _Float16 Ps[4 * 16 * 72];   // wave-private P tiles [16 q][64 key]
  const int tid = threadIdx.x;
  const int wave = tid >> 5, lane = tid & 31;
  const int l = lane & 15, hi = lane >> 4;
  const int qt = blockIdx.x;             // query tile of 64
  const int bh = blockIdx.y;             // b*16+h
  const int bb = bh >> 4, h = bh & 15;

  const _Float16* qbase = Q + (size_t)bh * 2048 * 64;
  const _Float16* kbase = Kg + (size_t)bh * 2048 * 64;
  const _Float16* vbase = Vt + (size_t)bh * 64 * 2048;
  _Float16* Pw = &Ps[wave * 16 * 72];

  // preload this wave's Q fragments (16 rows x 64 hd = 2 A-frags), kept all kernel
  v16h qf[2];
#pragma unroll
  for (int kk = 0; kk < 2; ++kk)
    qf[kk] = load_a_frag(qbase + (size_t)(qt * 64 + wave * 16) * 64 + kk * 32, 64);

  v8f o[4];
#pragma unroll
  for (int nt = 0; nt < 4; ++nt) o[nt] = (v8f)(0.0f);
  float mrow[8], lrow[8];
#pragma unroll
  for (int r = 0; r < 8; ++r) { mrow[r] = -1.0e30f; lrow[r] = 0.0f; }

  for (int kt = 0; kt < 32; ++kt) {
    __syncthreads();
    // stage K tile and V^T tile via async global->LDS DMA (16B per lane)
#pragma unroll
    for (int i = 0; i < 4; ++i) {
      int idx = tid + i * 128;           // 0..511 -> 64 rows x 8 groups of 8 halves
      int r = idx >> 3, c8 = idx & 7;
      async_copy_b128(&kbase[(size_t)(kt * 64 + r) * 64 + c8 * 8],
                      &Ks[r * 72 + c8 * 8]);
      async_copy_b128(&vbase[(size_t)r * 2048 + kt * 64 + c8 * 8],
                      &Vs[r * 72 + c8 * 8]);
    }
    wait_async0();                       // my wave's DMAs landed in LDS
    __syncthreads();                     // everyone's DMAs visible block-wide

    // scores S = Q @ K^T * 0.125 + mask   (4 n-tiles of 16 keys)
    v8f sc[4];
#pragma unroll
    for (int nt = 0; nt < 4; ++nt) {
      v8f s = (v8f)(0.0f);
#pragma unroll
      for (int kk = 0; kk < 2; ++kk) {
        v16h bfr = load_b_frag(&Ks[(nt * 16) * 72 + kk * 32], 72);
        s = WMMA_F16(qf[kk], bfr, s);
      }
      float mval = mask[bb * 2048 + kt * 64 + nt * 16 + l];
#pragma unroll
      for (int r = 0; r < 8; ++r) s[r] = s[r] * 0.125f + mval;
      sc[nt] = s;
    }

    // online softmax: row M = r + 8*hi lives in one 16-lane half of the wave,
    // so xor-butterfly over masks 8..1 reduces each half independently.
#pragma unroll
    for (int r = 0; r < 8; ++r) {
      float vmax = fmaxf(fmaxf(sc[0][r], sc[1][r]), fmaxf(sc[2][r], sc[3][r]));
#pragma unroll
      for (int mskk = 8; mskk >= 1; mskk >>= 1)
        vmax = fmaxf(vmax, __shfl_xor(vmax, mskk, 32));
      float mnew = fmaxf(mrow[r], vmax);
      float scale = __expf(mrow[r] - mnew);
      float psum = 0.0f;
#pragma unroll
      for (int nt = 0; nt < 4; ++nt) {
        float p = __expf(sc[nt][r] - mnew);
        sc[nt][r] = p;
        psum += p;
      }
#pragma unroll
      for (int mskk = 8; mskk >= 1; mskk >>= 1)
        psum += __shfl_xor(psum, mskk, 32);
      lrow[r] = lrow[r] * scale + psum;
      mrow[r] = mnew;
#pragma unroll
      for (int nt = 0; nt < 4; ++nt) o[nt][r] *= scale;
    }

    // reshape P: C-layout regs -> wave-private LDS [16 q][64 key] (A-layout source)
#pragma unroll
    for (int nt = 0; nt < 4; ++nt)
#pragma unroll
      for (int r = 0; r < 8; ++r)
        Pw[(r + 8 * hi) * 72 + nt * 16 + l] = (_Float16)sc[nt][r];
    asm volatile("s_wait_dscnt 0" ::: "memory");  // DS store -> DS load, same wave

    // O += P @ V  (A = P 16x64, B = V^T [hd][key] tiles)
    v16h pf[2];
#pragma unroll
    for (int kk = 0; kk < 2; ++kk) pf[kk] = load_a_frag(&Pw[kk * 32], 72);
#pragma unroll
    for (int nt = 0; nt < 4; ++nt)
#pragma unroll
      for (int kk = 0; kk < 2; ++kk) {
        v16h bfr = load_b_frag(&Vs[(nt * 16) * 72 + kk * 32], 72);
        o[nt] = WMMA_F16(pf[kk], bfr, o[nt]);
      }
  }

  // normalize and write ctx in [B,S,D] layout (f16) for the output projection
#pragma unroll
  for (int nt = 0; nt < 4; ++nt)
#pragma unroll
    for (int r = 0; r < 8; ++r) {
      int srow = qt * 64 + wave * 16 + r + 8 * hi;
      int n = h * 64 + nt * 16 + l;
      float v = o[nt][r] / lrow[r];
      ctx[((size_t)bb * 2048 + srow) * 1024 + n] = (_Float16)v;
    }
}

// ---------------------------------------------------------------------------
// Kernel 3: output projection.  out(f32) = ctx(f16)[4096,1024] @ Wo + bo
// ---------------------------------------------------------------------------
__global__ void __launch_bounds__(256)
oproj_gemm(const _Float16* __restrict__ A, const float* __restrict__ W,
           const float* __restrict__ bias, float* __restrict__ out) {
  __shared__ _Float16 As[128 * 40];
  __shared__ _Float16 Bt[128 * 40];
  const int tid = threadIdx.x;
  const int wave = tid >> 5, lane = tid & 31;
  const int waveM = wave >> 2, waveN = wave & 3;
  const int l = lane & 15, hi = lane >> 4;
  const int m0 = blockIdx.x * 128;
  const int n0 = blockIdx.y * 128;
  const int K = 1024, N = 1024;

  v8f acc[4][2];
#pragma unroll
  for (int mf = 0; mf < 4; ++mf)
#pragma unroll
    for (int nf = 0; nf < 2; ++nf) acc[mf][nf] = (v8f)(0.0f);

  for (int k0 = 0; k0 < K; k0 += 32) {
    __syncthreads();
    // stage A (already f16): async global->LDS DMA, 16B per lane
#pragma unroll
    for (int i = 0; i < 2; ++i) {
      int idx = tid + i * 256;           // 0..511 -> 128 rows x 4 groups of 8
      int r = idx >> 2, c8 = idx & 3;
      async_copy_b128(&A[(size_t)(m0 + r) * K + k0 + c8 * 8],
                      &As[r * 40 + c8 * 8]);
    }
    // stage Wo tile 32x128 transposed to [N][K]: float4 loads + convert
#pragma unroll
    for (int i = 0; i < 4; ++i) {
      int idx = tid + i * 256;
      int r = idx >> 5, c4 = idx & 31;
      float4 f = *(const float4*)&W[(size_t)(k0 + r) * N + n0 + c4 * 4];
      Bt[(c4 * 4 + 0) * 40 + r] = (_Float16)f.x;
      Bt[(c4 * 4 + 1) * 40 + r] = (_Float16)f.y;
      Bt[(c4 * 4 + 2) * 40 + r] = (_Float16)f.z;
      Bt[(c4 * 4 + 3) * 40 + r] = (_Float16)f.w;
    }
    wait_async0();
    __syncthreads();

    v16h bf[2];
#pragma unroll
    for (int nf = 0; nf < 2; ++nf)
      bf[nf] = load_b_frag(&Bt[(waveN * 32 + nf * 16) * 40], 40);
#pragma unroll
    for (int mf = 0; mf < 4; ++mf) {
      v16h af = load_a_frag(&As[(waveM * 64 + mf * 16) * 40], 40);
#pragma unroll
      for (int nf = 0; nf < 2; ++nf)
        acc[mf][nf] = WMMA_F16(af, bf[nf], acc[mf][nf]);
    }
  }

#pragma unroll
  for (int mf = 0; mf < 4; ++mf)
#pragma unroll
    for (int nf = 0; nf < 2; ++nf)
#pragma unroll
      for (int r = 0; r < 8; ++r) {
        int m = m0 + waveM * 64 + mf * 16 + r + 8 * hi;
        int n = n0 + waveN * 32 + nf * 16 + l;
        out[(size_t)m * 1024 + n] = acc[mf][nf][r] + bias[n];
      }
}

// ---------------------------------------------------------------------------
extern "C" void kernel_launch(void* const* d_in, const int* in_sizes, int n_in,
                              void* d_out, int out_size, void* d_ws, size_t ws_size,
                              hipStream_t stream) {
  (void)in_sizes; (void)n_in; (void)out_size; (void)ws_size;
  const float* x    = (const float*)d_in[0];
  const float* mask = (const float*)d_in[1];
  const float* Wq   = (const float*)d_in[2];
  const float* bq   = (const float*)d_in[3];
  const float* Wk   = (const float*)d_in[4];
  const float* bk   = (const float*)d_in[5];
  const float* Wv   = (const float*)d_in[6];
  const float* bv   = (const float*)d_in[7];
  const float* Wo   = (const float*)d_in[8];
  const float* bo   = (const float*)d_in[9];
  float* out = (float*)d_out;

  const size_t PER = (size_t)2 * 16 * 2048 * 64;  // 4,194,304 halves = 8 MB
  _Float16* qbuf  = (_Float16*)d_ws;
  _Float16* kbuf  = qbuf + PER;
  _Float16* vtbuf = kbuf + PER;
  _Float16* ctxb  = vtbuf + PER;

  dim3 gg(32, 8), bg(256);
  qkv_gemm<<<gg, bg, 0, stream>>>(x, Wq, bq, qbuf, 0);
  qkv_gemm<<<gg, bg, 0, stream>>>(x, Wk, bk, kbuf, 0);
  qkv_gemm<<<gg, bg, 0, stream>>>(x, Wv, bv, vtbuf, 1);
  attn_kernel<<<dim3(32, 32), dim3(128), 0, stream>>>(qbuf, kbuf, vtbuf, mask, ctxb);
  oproj_gemm<<<gg, bg, 0, stream>>>(ctxb, Wo, bo, out);
}
